// SO2EquivariantGraphAttentionWEdgesV2_31671088840830
// MI455X (gfx1250) — matrix operand order repaired and checked by reading, not compiled, not run on hardware
//
#include <hip/hip_runtime.h>
#include <hip/hip_bf16.h>

typedef __attribute__((ext_vector_type(16))) _Float16 v16h;
typedef __attribute__((ext_vector_type(8)))  float    v8f;

#define E_EDGES 40000
#define NN_NODES 5000
#define RBF 50

__constant__ int d_PERM[9]   = {0,2,6,1,5,3,7,4,8};
__constant__ int d_EXPAND[9] = {0,1,1,1,2,2,2,2,2};

// ---------------- WMMA fragment helpers (CDNA5 16x16x32 f16) ----------------
// Fragment-major storage: one 16x32 (A) or 32x16 (B) tile = 32 lanes x 16 f16,
// each lane's 16 values contiguous (32 B) -> 2x b128 loads per fragment.
__device__ inline v16h ld_frag_a(const _Float16* As, int tile) {
  return *(const v16h*)(As + ((size_t)tile << 9) + ((threadIdx.x & 31) << 4));
}
__device__ inline v16h ld_frag_b(const _Float16* bp) {
  return *(const v16h*)(bp + ((threadIdx.x & 31) << 4));
}
// swizzled f16 offset for scalar A staging (row r in 0..15, k within this mtile)
__device__ inline int sw_off(int r, int k) {
  int kt = k >> 5, kl = k & 31;
  int g = (kl >> 3) & 1;
  int i = (kl & 7) | ((kl & 16) >> 1);
  return (kt << 9) + ((r + (g << 4)) << 4) + i;
}
// swizzled f16 offset for one 8-f16 octet: ko = k/8 (16B-aligned destination)
__device__ inline int sw_oct_off(int r, int ko) {
  int kt = ko >> 2, koct = ko & 3;
  int g = koct & 1, i0 = (koct & 2) << 2;
  return (kt << 9) + ((r + (g << 4)) << 4) + i0;
}
__device__ inline void st_frag_lds(float* base, int ld, v8f c) {
  int lane = threadIdx.x & 31, n = lane & 15, g = lane >> 4;
#pragma unroll
  for (int r = 0; r < 8; ++r) base[(r + g*8)*ld + n] = c[r];
}
__device__ inline void st_frag_lds_h(_Float16* base, int ld, v8f c) {
  int lane = threadIdx.x & 31, n = lane & 15, g = lane >> 4;
#pragma unroll
  for (int r = 0; r < 8; ++r) base[(r + g*8)*ld + n] = (_Float16)c[r];
}

// ---- CDNA5 async global->LDS DMA (ASYNCcnt-tracked), 16 B per lane ----
__device__ inline void async_b128(void* lds_ptr, const void* gptr) {
  unsigned lds_off = (unsigned)(uintptr_t)lds_ptr;   // low 32 bits = LDS byte addr
  asm volatile("global_load_async_to_lds_b128 %0, %1, off"
               :: "v"(lds_off), "v"(gptr) : "memory");
}
__device__ inline void async_wait0() {
  asm volatile("s_wait_asynccnt 0x0" ::: "memory");
}

__device__ inline float silu_f(float x) { return x / (1.f + __expf(-x)); }

__device__ inline unsigned fenc(float f) {
  unsigned u = __float_as_uint(f);
  return (u & 0x80000000u) ? ~u : (u | 0x80000000u);
}
__device__ inline float fdec(unsigned u) {
  return __uint_as_float((u & 0x80000000u) ? (u & 0x7fffffffu) : ~u);
}

// ---------------- Weight pack: f32 KxN (row-major) -> fragment-major f16 ----------------
__global__ void k_pack(const float* __restrict__ src, _Float16* __restrict__ dst,
                       int K, int N, int Kvalid) {
  int KT = K >> 5, NT = N >> 4;
  int total8 = KT * NT * 64;
  int stride = gridDim.x * blockDim.x;
  for (int d8 = blockIdx.x*blockDim.x + threadIdx.x; d8 < total8; d8 += stride) {
    int d = d8 << 3;
    int i0   = d & 15;
    int lane = (d >> 4) & 31;
    int tile = d >> 9;
    int nt = tile % NT, kt = tile / NT;
    int n = nt*16 + (lane & 15);
    int kb = kt*32 + (lane >> 4)*8 + (i0 << 1);
    _Float16 tmp[8];
#pragma unroll
    for (int j = 0; j < 8; ++j) {
      int k = kb + j;
      tmp[j] = (_Float16)((k < Kvalid) ? src[(size_t)k*N + n] : 0.f);
    }
    *(uint4*)(dst + d) = *(const uint4*)tmp;
  }
}

// ---------------- Kernel 1: radial MLP (50 ->128 ->128 ->1536) ----------------
__device__ inline void ln_silu_16x128_sw(const float* H, _Float16* As,
                                         const float* g, const float* b,
                                         float* s_mu, float* s_rs) {
  int tid = threadIdx.x;
  if (tid < 16) {
    float s = 0.f, s2 = 0.f;
    for (int n = 0; n < 128; ++n) { float v = H[tid*128 + n]; s += v; s2 += v*v; }
    float mu = s * (1.f/128.f);
    float var = s2 * (1.f/128.f) - mu*mu;
    s_mu[tid] = mu; s_rs[tid] = rsqrtf(var + 1e-5f);
  }
  __syncthreads();
  for (int idx = tid; idx < 16*128; idx += 128) {
    int r = idx >> 7, n = idx & 127;
    float v = (H[idx] - s_mu[r]) * s_rs[r] * g[n] + b[n];
    As[sw_off(r, n)] = (_Float16)silu_f(v);
  }
  __syncthreads();
}

__global__ void __launch_bounds__(128) k_radial(
    const float* __restrict__ ed,
    const _Float16* __restrict__ pw0, const float* __restrict__ b0g,
    const float* __restrict__ g0g, const float* __restrict__ bb0g,
    const _Float16* __restrict__ pw1, const float* __restrict__ b1g,
    const float* __restrict__ g1g, const float* __restrict__ bb1g,
    const _Float16* __restrict__ pw2, const float* __restrict__ b2g,
    _Float16* __restrict__ rad16) {
  __shared__ __align__(32) _Float16 As[4*512];
  __shared__ float Hb[16*128];
  __shared__ float s_mu[16], s_rs[16];
  int tid = threadIdx.x, w = tid >> 5, lane = tid & 31;
  int nn = lane & 15, g = lane >> 4;
  int e0 = blockIdx.x * 16;

  for (int idx = tid; idx < 16*64; idx += 128) {
    int r = idx >> 6, k = idx & 63;
    float v = (k < RBF) ? ed[(size_t)(e0 + r)*RBF + k] : 0.f;
    As[sw_off(r, k)] = (_Float16)v;
  }
  __syncthreads();
  for (int t = w; t < 8; t += 4) {
    v8f acc = {};
#pragma unroll
    for (int kt = 0; kt < 2; ++kt) {
      v16h a = ld_frag_a(As, kt);
      v16h b = ld_frag_b(pw0 + ((size_t)(kt*8 + t) << 9));
      acc = __builtin_amdgcn_wmma_f32_16x16x32_f16(false, a, false, b, (short)0, acc, false, false);
    }
    float bias = b0g[t*16 + nn];
#pragma unroll
    for (int r = 0; r < 8; ++r) Hb[(r + g*8)*128 + t*16 + nn] = acc[r] + bias;
  }
  __syncthreads();
  ln_silu_16x128_sw(Hb, As, g0g, bb0g, s_mu, s_rs);
  for (int t = w; t < 8; t += 4) {
    v8f acc = {};
#pragma unroll
    for (int kt = 0; kt < 4; ++kt) {
      v16h a = ld_frag_a(As, kt);
      v16h b = ld_frag_b(pw1 + ((size_t)(kt*8 + t) << 9));
      acc = __builtin_amdgcn_wmma_f32_16x16x32_f16(false, a, false, b, (short)0, acc, false, false);
    }
    float bias = b1g[t*16 + nn];
#pragma unroll
    for (int r = 0; r < 8; ++r) Hb[(r + g*8)*128 + t*16 + nn] = acc[r] + bias;
  }
  __syncthreads();
  ln_silu_16x128_sw(Hb, As, g1g, bb1g, s_mu, s_rs);
  for (int t = w; t < 96; t += 4) {
    v8f acc = {};
#pragma unroll
    for (int kt = 0; kt < 4; ++kt) {
      v16h a = ld_frag_a(As, kt);
      v16h b = ld_frag_b(pw2 + ((size_t)(kt*96 + t) << 9));
      acc = __builtin_amdgcn_wmma_f32_16x16x32_f16(false, a, false, b, (short)0, acc, false, false);
    }
    float bias = b2g[t*16 + nn];
#pragma unroll
    for (int r = 0; r < 8; ++r)
      rad16[(size_t)(e0 + r + g*8)*1536 + t*16 + nn] = (_Float16)(acc[r] + bias);
  }
}

// ------- Kernel 2: gather + wigner rotate + rad scale -> xm1 (E x 2304 f16) -------
__global__ void __launch_bounds__(256) k_rot(
    const float* __restrict__ x, const int* __restrict__ ei,
    const float* __restrict__ wig, const _Float16* __restrict__ rad16,
    _Float16* __restrict__ xm1) {
  __shared__ float wg[81];
  int e = blockIdx.x, tid = threadIdx.x;
  if (tid < 81) wg[tid] = wig[(size_t)e*81 + tid];
  __syncthreads();
  int src = ei[e], tgt = ei[E_EDGES + e];
  int c = tid;
  const float* xb = (c < 128) ? (x + (size_t)src*1152 + c)
                              : (x + (size_t)tgt*1152 + (c - 128));
  float m[9];
#pragma unroll
  for (int j = 0; j < 9; ++j) m[j] = xb[j*128];
  float rot[9];
#pragma unroll
  for (int i = 0; i < 9; ++i) {
    float s = 0.f;
#pragma unroll
    for (int j = 0; j < 9; ++j) s += wg[i*9 + j] * m[j];
    rot[i] = s;
  }
  const _Float16* rd = rad16 + (size_t)e*1536;
  _Float16* o = xm1 + (size_t)e*2304;
  o[c]       = (_Float16)(rot[0] * (float)rd[c]);
  o[256 + c] = (_Float16)(rot[2] * (float)rd[256 + c]);
  o[512 + c] = (_Float16)(rot[6] * (float)rd[512 + c]);
  float r1 = (float)rd[768 + c], r2 = (float)rd[1024 + c];
  o[768 + c]       = (_Float16)(rot[1] * r1);
  o[768 + 256 + c] = (_Float16)(rot[5] * r2);
  o[768 + 512 + c] = (_Float16)(rot[3] * r1);
  o[768 + 768 + c] = (_Float16)(rot[7] * r2);
  float r3 = (float)rd[1280 + c];
  o[1792 + c]       = (_Float16)(rot[4] * r3);
  o[1792 + 256 + c] = (_Float16)(rot[8] * r3);
}

// ---------------- Kernel 3: SO2 conv 1 (WMMA GEMMs, async LDS staging) ----------------
__global__ void __launch_bounds__(256) k_conv1(
    const _Float16* __restrict__ xm1,
    const _Float16* __restrict__ pw0, const float* __restrict__ b0,
    const _Float16* __restrict__ pw1m, const _Float16* __restrict__ pw2m,
    float* __restrict__ xa, float* __restrict__ msg1) {
  __shared__ __align__(32) _Float16 As[32*512];
  __shared__ float Y[32*256];
  int tid = threadIdx.x, w = tid >> 5, lane = tid & 31;
  int nn = lane & 15, g = lane >> 4;
  int e0 = blockIdx.x * 16;

  for (int idx = tid; idx < 16*96; idx += 256) {
    int r = idx / 96, ko = idx % 96;
    async_b128(As + sw_oct_off(r, ko), xm1 + (size_t)(e0 + r)*2304 + ko*8);
  }
  async_wait0();
  __syncthreads();
  for (int t = w; t < 32; t += 8) {
    v8f acc = {};
    for (int kt = 0; kt < 24; ++kt) {
      v16h a = ld_frag_a(As, kt);
      v16h b = ld_frag_b(pw0 + ((size_t)(kt*32 + t) << 9));
      acc = __builtin_amdgcn_wmma_f32_16x16x32_f16(false, a, false, b, (short)0, acc, false, false);
    }
    int n = t*16 + nn;
    float bias = b0[n];
#pragma unroll
    for (int r = 0; r < 8; ++r) {
      int e = e0 + r + g*8;
      float val = acc[r] + bias;
      if (n < 320) xa[(size_t)e*320 + n] = val;
      else {
        int p = (n - 320) >> 6, co = (n - 320) & 63;
        msg1[((size_t)e*9 + d_PERM[p])*64 + co] = val;
      }
    }
  }
  __syncthreads();
  for (int idx = tid; idx < 32*64; idx += 256) {
    int rr = idx / 64, ko = idx % 64;
    int mt = rr >> 4, m = rr & 15;
    async_b128(As + mt*16*512 + sw_oct_off(m, ko),
               xm1 + (size_t)(e0 + (rr >> 1))*2304 + 768 + (rr & 1)*512 + ko*8);
  }
  async_wait0();
  __syncthreads();
  for (int t = w; t < 32; t += 8) {
    int mt = t >> 4, nt = t & 15;
    v8f acc = {};
    for (int kt = 0; kt < 16; ++kt) {
      v16h a = ld_frag_a(As, mt*16 + kt);
      v16h b = ld_frag_b(pw1m + ((size_t)(kt*16 + nt) << 9));
      acc = __builtin_amdgcn_wmma_f32_16x16x32_f16(false, a, false, b, (short)0, acc, false, false);
    }
    st_frag_lds(Y + mt*16*256 + nt*16, 256, acc);
  }
  __syncthreads();
  for (int idx = tid; idx < 16*2*128; idx += 256) {
    int el = idx >> 8, rem = idx & 255, rr = rem >> 7, c = rem & 127;
    float v;
    if (rr == 0) v = Y[(2*el)*256 + c]   - Y[(2*el+1)*256 + 128 + c];
    else         v = Y[(2*el+1)*256 + c] + Y[(2*el)*256 + 128 + c];
    int p = 3 + rr*2 + (c >= 64 ? 1 : 0), co = c & 63;
    msg1[((size_t)(e0 + el)*9 + d_PERM[p])*64 + co] = v;
  }
  __syncthreads();
  for (int idx = tid; idx < 32*32; idx += 256) {
    int rr = idx / 32, ko = idx % 32;
    int mt = rr >> 4, m = rr & 15;
    async_b128(As + mt*8*512 + sw_oct_off(m, ko),
               xm1 + (size_t)(e0 + (rr >> 1))*2304 + 1792 + (rr & 1)*256 + ko*8);
  }
  async_wait0();
  __syncthreads();
  for (int t = w; t < 16; t += 8) {
    int mt = t >> 3, nt = t & 7;
    v8f acc = {};
    for (int kt = 0; kt < 8; ++kt) {
      v16h a = ld_frag_a(As, mt*8 + kt);
      v16h b = ld_frag_b(pw2m + ((size_t)(kt*8 + nt) << 9));
      acc = __builtin_amdgcn_wmma_f32_16x16x32_f16(false, a, false, b, (short)0, acc, false, false);
    }
    st_frag_lds(Y + mt*16*128 + nt*16, 128, acc);
  }
  __syncthreads();
  for (int idx = tid; idx < 16*2*64; idx += 256) {
    int el = idx >> 7, rem = idx & 127, rr = rem >> 6, c = rem & 63;
    float v;
    if (rr == 0) v = Y[(2*el)*128 + c]   - Y[(2*el+1)*128 + 64 + c];
    else         v = Y[(2*el+1)*128 + c] + Y[(2*el)*128 + 64 + c];
    msg1[((size_t)(e0 + el)*9 + d_PERM[7 + rr])*64 + c] = v;
  }
}

// ---------------- Kernel 4: S2 grid nonlinearity -> xm2 ----------------
__global__ void __launch_bounds__(256) k_grid(
    const float* __restrict__ msg1, const float* __restrict__ xa,
    const float* __restrict__ to_grid, const float* __restrict__ from_grid,
    _Float16* __restrict__ xm2) {
  __shared__ float tg[42*9], fg[42*9];
  int tid = threadIdx.x;
  for (int i = tid; i < 42*9; i += 256) { tg[i] = to_grid[i]; fg[i] = from_grid[i]; }
  __syncthreads();
  int e = blockIdx.x*4 + (tid >> 6);
  int c = tid & 63;
  float v[9], s2[9];
#pragma unroll
  for (int i = 0; i < 9; ++i) { v[i] = msg1[((size_t)e*9 + i)*64 + c]; s2[i] = 0.f; }
  for (int gidx = 0; gidx < 42; ++gidx) {
    float t = 0.f;
#pragma unroll
    for (int i = 0; i < 9; ++i) t += tg[gidx*9 + i] * v[i];
    t = silu_f(t);
#pragma unroll
    for (int i = 0; i < 9; ++i) s2[i] += fg[gidx*9 + i] * t;
  }
  float m2[9];
  m2[0] = silu_f(xa[(size_t)e*320 + 256 + c]);
#pragma unroll
  for (int i = 1; i < 9; ++i) m2[i] = s2[i];
  _Float16* o = xm2 + (size_t)e*576;
  o[c]        = (_Float16)m2[0];
  o[64 + c]   = (_Float16)m2[2];
  o[128 + c]  = (_Float16)m2[6];
  o[192 + c]       = (_Float16)m2[1];
  o[192 + 64 + c]  = (_Float16)m2[5];
  o[192 + 128 + c] = (_Float16)m2[3];
  o[192 + 192 + c] = (_Float16)m2[7];
  o[448 + c]       = (_Float16)m2[4];
  o[448 + 64 + c]  = (_Float16)m2[8];
}

// ---------------- Kernel 5: SO2 conv 2 (WMMA GEMMs, async LDS staging) ----------------
__global__ void __launch_bounds__(256) k_conv2(
    const _Float16* __restrict__ xm2,
    const _Float16* __restrict__ pw0, const float* __restrict__ b0,
    const _Float16* __restrict__ pw1m, const _Float16* __restrict__ pw2m,
    float* __restrict__ y2) {
  __shared__ __align__(32) _Float16 As[8*512];
  __shared__ __align__(32) _Float16 Yh[32*512];
  int tid = threadIdx.x, w = tid >> 5, lane = tid & 31;
  int nn = lane & 15, g = lane >> 4;
  int e0 = blockIdx.x * 16;

  for (int idx = tid; idx < 16*24; idx += 256) {
    int r = idx / 24, ko = idx % 24;
    async_b128(As + sw_oct_off(r, ko), xm2 + (size_t)(e0 + r)*576 + ko*8);
  }
  async_wait0();
  __syncthreads();
  for (int t = w; t < 24; t += 8) {
    v8f acc = {};
#pragma unroll
    for (int kt = 0; kt < 6; ++kt) {
      v16h a = ld_frag_a(As, kt);
      v16h b = ld_frag_b(pw0 + ((size_t)(kt*24 + t) << 9));
      acc = __builtin_amdgcn_wmma_f32_16x16x32_f16(false, a, false, b, (short)0, acc, false, false);
    }
    int n = t*16 + nn;
    float bias = b0[n];
    int p = n >> 7, co = n & 127;
#pragma unroll
    for (int r = 0; r < 8; ++r) {
      int e = e0 + r + g*8;
      y2[((size_t)e*9 + d_PERM[p])*128 + co] = acc[r] + bias;
    }
  }
  __syncthreads();
  for (int idx = tid; idx < 32*16; idx += 256) {
    int rr = idx / 16, ko = idx % 16;
    int mt = rr >> 4, m = rr & 15;
    async_b128(As + mt*4*512 + sw_oct_off(m, ko),
               xm2 + (size_t)(e0 + (rr >> 1))*576 + 192 + (rr & 1)*128 + ko*8);
  }
  async_wait0();
  __syncthreads();
  for (int t = w; t < 64; t += 8) {
    int mt = t >> 5, nt = t & 31;
    v8f acc = {};
#pragma unroll
    for (int kt = 0; kt < 4; ++kt) {
      v16h a = ld_frag_a(As, mt*4 + kt);
      v16h b = ld_frag_b(pw1m + ((size_t)(kt*32 + nt) << 9));
      acc = __builtin_amdgcn_wmma_f32_16x16x32_f16(false, a, false, b, (short)0, acc, false, false);
    }
    st_frag_lds_h(Yh + mt*16*512 + nt*16, 512, acc);
  }
  __syncthreads();
  for (int idx = tid; idx < 16*2*256; idx += 256) {
    int el = idx >> 9, rem = idx & 511, rr = rem >> 8, c = rem & 255;
    float v;
    if (rr == 0) v = (float)Yh[(2*el)*512 + c]   - (float)Yh[(2*el+1)*512 + 256 + c];
    else         v = (float)Yh[(2*el+1)*512 + c] + (float)Yh[(2*el)*512 + 256 + c];
    int p = 3 + rr*2 + (c >= 128 ? 1 : 0), co = c & 127;
    y2[((size_t)(e0 + el)*9 + d_PERM[p])*128 + co] = v;
  }
  __syncthreads();
  for (int idx = tid; idx < 32*8; idx += 256) {
    int rr = idx / 8, ko = idx % 8;
    int mt = rr >> 4, m = rr & 15;
    async_b128(As + mt*2*512 + sw_oct_off(m, ko),
               xm2 + (size_t)(e0 + (rr >> 1))*576 + 448 + (rr & 1)*64 + ko*8);
  }
  async_wait0();
  __syncthreads();
  for (int t = w; t < 32; t += 8) {
    int mt = t >> 4, nt = t & 15;
    v8f acc = {};
#pragma unroll
    for (int kt = 0; kt < 2; ++kt) {
      v16h a = ld_frag_a(As, mt*2 + kt);
      v16h b = ld_frag_b(pw2m + ((size_t)(kt*16 + nt) << 9));
      acc = __builtin_amdgcn_wmma_f32_16x16x32_f16(false, a, false, b, (short)0, acc, false, false);
    }
    st_frag_lds_h(Yh + mt*16*256 + nt*16, 256, acc);
  }
  __syncthreads();
  for (int idx = tid; idx < 16*2*128; idx += 256) {
    int el = idx >> 8, rem = idx & 255, rr = rem >> 7, c = rem & 127;
    float v;
    if (rr == 0) v = (float)Yh[(2*el)*256 + c]   - (float)Yh[(2*el+1)*256 + 128 + c];
    else         v = (float)Yh[(2*el+1)*256 + c] + (float)Yh[(2*el)*256 + 128 + c];
    y2[((size_t)(e0 + el)*9 + d_PERM[7 + rr])*128 + c] = v;
  }
}

// ---------------- Attention + scatter kernels ----------------
__global__ void k_init(float* node, unsigned* amax, float* denom) {
  int i = blockIdx.x*blockDim.x + threadIdx.x;
  if (i < NN_NODES*1152) node[i] = 0.f;
  if (i < NN_NODES*8) { amax[i] = 0x007FFFFFu; denom[i] = 0.f; }
}

__global__ void k_alpha(const float* __restrict__ xa, const int* __restrict__ ei,
                        const float* __restrict__ ag, const float* __restrict__ ab,
                        const float* __restrict__ adot,
                        float* __restrict__ aex, unsigned* __restrict__ amax) {
  int i = blockIdx.x*blockDim.x + threadIdx.x;
  if (i >= E_EDGES*8) return;
  int e = i >> 3, h = i & 7;
  const float* xv = xa + (size_t)e*320 + h*32;
  float s = 0.f, s2 = 0.f;
#pragma unroll
  for (int k = 0; k < 32; ++k) { float v = xv[k]; s += v; s2 += v*v; }
  float mu = s * (1.f/32.f);
  float var = s2 * (1.f/32.f) - mu*mu;
  float rs = rsqrtf(var + 1e-5f);
  float acc = 0.f;
#pragma unroll
  for (int k = 0; k < 32; ++k) {
    float v = (xv[k] - mu) * rs * ag[k] + ab[k];
    float sg = 1.f / (1.f + __expf(-v));
    float sl = 0.6f*v + 0.4f*v*(2.f*sg - 1.f);
    acc += sl * adot[h*32 + k];
  }
  aex[i] = acc;
  int tgt = ei[E_EDGES + e];
  atomicMax(&amax[tgt*8 + h], fenc(acc));
}

__global__ void k_exp(const int* __restrict__ ei, float* __restrict__ aex,
                      const unsigned* __restrict__ amax, float* __restrict__ denom) {
  int i = blockIdx.x*blockDim.x + threadIdx.x;
  if (i >= E_EDGES*8) return;
  int e = i >> 3, h = i & 7;
  int tgt = ei[E_EDGES + e];
  float mx = fdec(amax[tgt*8 + h]);
  if (!(fabsf(mx) < 3.0e38f)) mx = 0.f;
  float ex = __expf(aex[i] - mx);
  aex[i] = ex;
  atomicAdd(&denom[tgt*8 + h], ex);
}

__global__ void __launch_bounds__(128) k_scatter(
    const float* __restrict__ y2, const float* __restrict__ winv,
    const int* __restrict__ ei, const float* __restrict__ aex,
    const float* __restrict__ denom, float* __restrict__ node) {
  __shared__ float wi[81];
  __shared__ float f[8];
  int e = blockIdx.x, tid = threadIdx.x;
  int tgt = ei[E_EDGES + e];
  if (tid < 81) wi[tid] = winv[(size_t)e*81 + tid];
  if (tid < 8)  f[tid] = aex[(size_t)e*8 + tid] / (denom[tgt*8 + tid] + 1e-16f);
  __syncthreads();
  int c = tid;
  float fac = f[c >> 4];
  float v[9];
#pragma unroll
  for (int j = 0; j < 9; ++j) v[j] = y2[((size_t)e*9 + j)*128 + c] * fac;
#pragma unroll
  for (int i = 0; i < 9; ++i) {
    float r = 0.f;
#pragma unroll
    for (int j = 0; j < 9; ++j) r += wi[i*9 + j] * v[j];
    atomicAdd(&node[((size_t)tgt*9 + i)*128 + c], r);
  }
}

// ---------------- Kernel: output projection (WMMA) ----------------
__global__ void __launch_bounds__(256) k_proj(
    const float* __restrict__ node, const _Float16* __restrict__ pwp,
    const float* __restrict__ pb, float* __restrict__ out) {
  __shared__ __align__(32) _Float16 As[4*512];
  int tid = threadIdx.x, w = tid >> 5, lane = tid & 31;
  int nn = lane & 15, g = lane >> 4;
  int n0 = blockIdx.x * 16, m = blockIdx.y;
  int l = d_EXPAND[m];
  for (int idx = tid; idx < 16*128; idx += 256) {
    int r = idx >> 7, k = idx & 127;
    float v = (n0 + r < NN_NODES) ? node[((size_t)(n0 + r)*9 + m)*128 + k] : 0.f;
    As[sw_off(r, k)] = (_Float16)v;
  }
  __syncthreads();
  v8f acc = {};
#pragma unroll
  for (int kt = 0; kt < 4; ++kt) {
    v16h a = ld_frag_a(As, kt);
    v16h b = ld_frag_b(pwp + (size_t)l*16384 + ((size_t)(kt*8 + w) << 9));
    acc = __builtin_amdgcn_wmma_f32_16x16x32_f16(false, a, false, b, (short)0, acc, false, false);
  }
  int ncol = w*16 + nn;
  float bias = (m == 0) ? pb[ncol] : 0.f;
#pragma unroll
  for (int r = 0; r < 8; ++r) {
    int row = n0 + r + g*8;
    if (row < NN_NODES) out[((size_t)row*9 + m)*128 + ncol] = acc[r] + bias;
  }
}

// ---------------- launch ----------------
extern "C" void kernel_launch(void* const* d_in, const int* in_sizes, int n_in,
                              void* d_out, int out_size, void* d_ws, size_t ws_size,
                              hipStream_t stream) {
  (void)in_sizes; (void)n_in; (void)out_size; (void)ws_size;
  const float* x         = (const float*)d_in[0];
  const float* ed        = (const float*)d_in[1];
  const int*   ei        = (const int*)  d_in[2];
  const float* wig       = (const float*)d_in[3];
  const float* wigi      = (const float*)d_in[4];
  const float* rad_w0    = (const float*)d_in[5];
  const float* rad_b0    = (const float*)d_in[6];
  const float* rad_g0    = (const float*)d_in[7];
  const float* rad_bb0   = (const float*)d_in[8];
  const float* rad_w1    = (const float*)d_in[9];
  const float* rad_b1    = (const float*)d_in[10];
  const float* rad_g1    = (const float*)d_in[11];
  const float* rad_bb1   = (const float*)d_in[12];
  const float* rad_w2    = (const float*)d_in[13];
  const float* rad_b2    = (const float*)d_in[14];
  const float* c1_w0     = (const float*)d_in[15];
  const float* c1_b0     = (const float*)d_in[16];
  const float* c1_w1m    = (const float*)d_in[17];
  const float* c1_w2m    = (const float*)d_in[18];
  const float* to_grid   = (const float*)d_in[19];
  const float* from_grid = (const float*)d_in[20];
  const float* c2_w0     = (const float*)d_in[21];
  const float* c2_b0     = (const float*)d_in[22];
  const float* c2_w1m    = (const float*)d_in[23];
  const float* c2_w2m    = (const float*)d_in[24];
  const float* alpha_g   = (const float*)d_in[25];
  const float* alpha_b   = (const float*)d_in[26];
  const float* alpha_dot = (const float*)d_in[27];
  const float* proj_w    = (const float*)d_in[28];
  const float* proj_b    = (const float*)d_in[29];
  float* out = (float*)d_out;

  char* wsp = (char*)d_ws;
  _Float16* rad16 = (_Float16*)wsp; wsp += (size_t)E_EDGES*1536*sizeof(_Float16);
  _Float16* xm1   = (_Float16*)wsp; wsp += (size_t)E_EDGES*2304*sizeof(_Float16);
  float* xa       = (float*)wsp;    wsp += (size_t)E_EDGES*320*sizeof(float);
  float* msg1     = (float*)wsp;    wsp += (size_t)E_EDGES*576*sizeof(float);
  _Float16* xm2   = (_Float16*)wsp; wsp += (size_t)E_EDGES*576*sizeof(_Float16);
  float* y2       = (float*)wsp;    wsp += (size_t)E_EDGES*1152*sizeof(float);
  float* aex      = (float*)wsp;    wsp += (size_t)E_EDGES*8*sizeof(float);
  unsigned* amax  = (unsigned*)wsp; wsp += (size_t)NN_NODES*8*sizeof(unsigned);
  float* denom    = (float*)wsp;    wsp += (size_t)NN_NODES*8*sizeof(float);
  float* node     = (float*)wsp;    wsp += (size_t)NN_NODES*1152*sizeof(float);
  _Float16* p_rad0 = (_Float16*)wsp; wsp += (size_t)64*128*2;
  _Float16* p_rad1 = (_Float16*)wsp; wsp += (size_t)128*128*2;
  _Float16* p_rad2 = (_Float16*)wsp; wsp += (size_t)128*1536*2;
  _Float16* p_c1w0 = (_Float16*)wsp; wsp += (size_t)768*512*2;
  _Float16* p_c1w1 = (_Float16*)wsp; wsp += (size_t)512*256*2;
  _Float16* p_c1w2 = (_Float16*)wsp; wsp += (size_t)256*128*2;
  _Float16* p_c2w0 = (_Float16*)wsp; wsp += (size_t)192*384*2;
  _Float16* p_c2w1 = (_Float16*)wsp; wsp += (size_t)128*512*2;
  _Float16* p_c2w2 = (_Float16*)wsp; wsp += (size_t)64*256*2;
  _Float16* p_proj = (_Float16*)wsp; wsp += (size_t)3*128*128*2;

#define PACK(src, dst, K, N, KV) \
  k_pack<<<(((K)*(N)/8) + 255)/256, 256, 0, stream>>>(src, dst, K, N, KV)
  PACK(rad_w0, p_rad0, 64, 128, 50);
  PACK(rad_w1, p_rad1, 128, 128, 128);
  PACK(rad_w2, p_rad2, 128, 1536, 128);
  PACK(c1_w0,  p_c1w0, 768, 512, 768);
  PACK(c1_w1m, p_c1w1, 512, 256, 512);
  PACK(c1_w2m, p_c1w2, 256, 128, 256);
  PACK(c2_w0,  p_c2w0, 192, 384, 192);
  PACK(c2_w1m, p_c2w1, 128, 512, 128);
  PACK(c2_w2m, p_c2w2, 64, 256, 64);
  PACK(proj_w,         p_proj,         128, 128, 128);
  PACK(proj_w + 16384, p_proj + 16384, 128, 128, 128);
  PACK(proj_w + 32768, p_proj + 32768, 128, 128, 128);
#undef PACK

  k_radial<<<E_EDGES/16, 128, 0, stream>>>(ed, p_rad0, rad_b0, rad_g0, rad_bb0,
      p_rad1, rad_b1, rad_g1, rad_bb1, p_rad2, rad_b2, rad16);
  k_rot<<<E_EDGES, 256, 0, stream>>>(x, ei, wig, rad16, xm1);
  k_conv1<<<E_EDGES/16, 256, 0, stream>>>(xm1, p_c1w0, c1_b0, p_c1w1, p_c1w2, xa, msg1);
  k_grid<<<E_EDGES/4, 256, 0, stream>>>(msg1, xa, to_grid, from_grid, xm2);
  k_conv2<<<E_EDGES/16, 256, 0, stream>>>(xm2, p_c2w0, c2_b0, p_c2w1, p_c2w2, y2);
  k_init<<<(NN_NODES*1152 + 255)/256, 256, 0, stream>>>(node, amax, denom);
  k_alpha<<<(E_EDGES*8 + 255)/256, 256, 0, stream>>>(xa, ei, alpha_g, alpha_b, alpha_dot, aex, amax);
  k_exp<<<(E_EDGES*8 + 255)/256, 256, 0, stream>>>(ei, aex, amax, denom);
  k_scatter<<<E_EDGES, 128, 0, stream>>>(y2, wigi, ei, aex, denom, node);
  dim3 pg((NN_NODES + 15)/16, 9);
  k_proj<<<pg, 256, 0, stream>>>(node, p_proj, proj_b, out);
}
